// PASTBlock_44444321579634
// MI455X (gfx1250) — matrix operand order repaired and checked by reading, not compiled, not run on hardware
//
#include <hip/hip_runtime.h>
#include <hip/hip_bf16.h>
#include <hip/hip_fp16.h>

typedef __attribute__((ext_vector_type(16))) _Float16 v16h;
typedef __attribute__((ext_vector_type(8)))  _Float16 v8h;
typedef __attribute__((ext_vector_type(8)))  float    v8f;

#define C_DIM   256
#define NHEADS  8
#define DHEAD   32
#define NTOK    49
#define NKV     52
#define SHIFT3  3
#define HWDIM   112
#define LTOK    12544      // 112*112
#define NW_B    256        // windows per batch (16x16)
#define BATCH   4
#define NW_TOT  1024       // BATCH * NW_B
#define MROWS   50176      // BATCH * LTOK  == NW_TOT*49
#define HID     1024
#define QSCALE  0.17677669529663687f   // 32^-0.5

// ---------------------------------------------------------------------------
// WMMA fragment loads (CDNA5 16-bit layouts, wave32)
// A 16x32 f16: lane l -> row m=l&15; VGPR0-3 hold K=hi*8..hi*8+7,
//              VGPR4-7 hold K=16+hi*8.. (hi = l>>4)
// B 32x16 f16 (staged n-major, K contiguous): lane l -> col n=l&15,
//              VGPR0-3 hold K=hi*8..hi*8+7, VGPR4-7 hold K=16+hi*8.. (same order as A)
// C/D f32 16x16: lane l -> col n=l&15, VGPR j -> row m=(l>>4)*8+j
// ---------------------------------------------------------------------------
__device__ __forceinline__ v16h frag_a(const _Float16* t, int lane, int pitch) {
    const int m  = lane & 15;
    const int hi = (lane >> 4) << 3;           // 0 or 8
    const _Float16* p = t + m * pitch;
    v8h lo = *(const v8h*)(p + hi);
    v8h hh = *(const v8h*)(p + 16 + hi);
    v16h r;
#pragma unroll
    for (int i = 0; i < 8; ++i) { r[i] = lo[i]; r[8 + i] = hh[i]; }
    return r;
}

__device__ __forceinline__ v16h frag_b(const _Float16* t, int lane, int pitch) {
    const int n  = lane & 15;
    const int hi = (lane >> 4) << 3;           // 0 or 8
    const _Float16* p = t + n * pitch + hi;
    v8h lo = *(const v8h*)(p);
    v8h hh = *(const v8h*)(p + 16);
    v16h r;
#pragma unroll
    for (int i = 0; i < 8; ++i) { r[i] = lo[i]; r[8 + i] = hh[i]; }
    return r;
}

__device__ __forceinline__ v8f wmma_f16(v16h a, v16h b, v8f c) {
    return __builtin_amdgcn_wmma_f32_16x16x32_f16(
        /*neg_a=*/false, a, /*neg_b=*/false, b,
        /*c_mod=*/(short)0, c, /*reuse_a=*/false, /*reuse_b=*/false);
}

// ---------------------------------------------------------------------------
// Shared 64(M) x 128(N) block mainloop, double-buffered LDS filled by
// CDNA5 async global->LDS DMA (GLOBAL_LOAD_ASYNC_TO_LDS_B128, ASYNCcnt).
// A: global row-major (lda halves), B: global row-major (N,K) (ldb halves).
// 8 waves: wave w -> mt=w>>1, 4 n-tiles at (w&1)*4.
// ---------------------------------------------------------------------------
__device__ __forceinline__ void wmma_block_64x128(
    const _Float16* __restrict__ Ag, int lda,
    const _Float16* __restrict__ Bg, int ldb,
    int K, v8f acc[4])
{
    __shared__ __attribute__((aligned(16))) _Float16 sA[2][64 * 40];
    __shared__ __attribute__((aligned(16))) _Float16 sB[2][128 * 40];
    const int tid  = threadIdx.x;
    const int lane = tid & 31;
    const int w8   = tid >> 5;
    const int mt   = w8 >> 1;
    const int ntb  = (w8 & 1) * 4;
    const v8f z = {0.f,0.f,0.f,0.f,0.f,0.f,0.f,0.f};
#pragma unroll
    for (int i = 0; i < 4; ++i) acc[i] = z;

    // per-thread staging slots: A = 256 chunks of 8 halves, B = 512 chunks
    const int rowA = tid >> 2, offA = (tid & 3) * 8;
    const int rB0  = tid >> 2, rB1 = (tid >> 2) + 64, oB = (tid & 3) * 8;

    auto issue = [&](int buf, int kk) {
        unsigned la  = (unsigned)(size_t)&sA[buf][rowA * 40 + offA];
        unsigned lb0 = (unsigned)(size_t)&sB[buf][rB0 * 40 + oB];
        unsigned lb1 = (unsigned)(size_t)&sB[buf][rB1 * 40 + oB];
        unsigned long long ga  = (unsigned long long)&Ag[(size_t)rowA * lda + kk + offA];
        unsigned long long gb0 = (unsigned long long)&Bg[(size_t)rB0 * ldb + kk + oB];
        unsigned long long gb1 = (unsigned long long)&Bg[(size_t)rB1 * ldb + kk + oB];
        asm volatile("global_load_async_to_lds_b128 %0, %1, off"
                     :: "v"(la), "v"(ga) : "memory");
        asm volatile("global_load_async_to_lds_b128 %0, %1, off"
                     :: "v"(lb0), "v"(gb0) : "memory");
        asm volatile("global_load_async_to_lds_b128 %0, %1, off"
                     :: "v"(lb1), "v"(gb1) : "memory");
    };

    issue(0, 0);
    const int nsteps = K >> 5;
    for (int s = 0; s < nsteps; ++s) {
        const int cur = s & 1;
        if (s + 1 < nsteps) {
            issue(cur ^ 1, (s + 1) << 5);
            // async loads complete in order: <=3 outstanding => stage s resident
            asm volatile("s_wait_asynccnt 0x3" ::: "memory");
        } else {
            asm volatile("s_wait_asynccnt 0x0" ::: "memory");
        }
        __syncthreads();
        v16h a = frag_a(&sA[cur][mt * 16 * 40], lane, 40);
#pragma unroll
        for (int t = 0; t < 4; ++t) {
            v16h b = frag_b(&sB[cur][(ntb + t) * 16 * 40], lane, 40);
            acc[t] = wmma_f16(a, b, acc[t]);
        }
        __syncthreads();
    }
}

// ---------------------------------------------------------------------------
// f32 -> f16 convert
// ---------------------------------------------------------------------------
__global__ void k_cvt(const float* __restrict__ s, _Float16* __restrict__ d, int n) {
    int i = blockIdx.x * 256 + threadIdx.x;
    if (i < n) d[i] = (_Float16)s[i];
}

// ---------------------------------------------------------------------------
// Pooled priors (uk, fg, bg) + LN1 on them -> prior_h (B,3,256) f16
// ---------------------------------------------------------------------------
__global__ void k_priors(const float* __restrict__ x,
                         const float* __restrict__ auk,
                         const float* __restrict__ afg,
                         const float* __restrict__ abg,
                         const float* __restrict__ g, const float* __restrict__ bb,
                         _Float16* __restrict__ prior_h)
{
    __shared__ float red[256];
    __shared__ float s_inv[3];
    __shared__ float pool[3 * 256];
    const int b = blockIdx.x, t = threadIdx.x;
    const float* A[3] = { auk + (size_t)b * LTOK, afg + (size_t)b * LTOK, abg + (size_t)b * LTOK };

    for (int pr = 0; pr < 3; ++pr) {
        float s = 0.f;
        for (int p = t; p < LTOK; p += 256) s += A[pr][p];
        red[t] = s; __syncthreads();
        for (int st = 128; st > 0; st >>= 1) { if (t < st) red[t] += red[t + st]; __syncthreads(); }
        if (t == 0) s_inv[pr] = 1.f / (red[0] + 1e-9f);
        __syncthreads();
    }
    float acc0 = 0.f, acc1 = 0.f, acc2 = 0.f;
    const float* xb = x + (size_t)b * LTOK * C_DIM;
    for (int p = 0; p < LTOK; ++p) {
        float xv = xb[(size_t)p * C_DIM + t];
        acc0 += A[0][p] * xv; acc1 += A[1][p] * xv; acc2 += A[2][p] * xv;
    }
    pool[0 * 256 + t] = acc0 * s_inv[0];
    pool[1 * 256 + t] = acc1 * s_inv[1];
    pool[2 * 256 + t] = acc2 * s_inv[2];
    __syncthreads();
    for (int pr = 0; pr < 3; ++pr) {
        float v = pool[pr * 256 + t];
        red[t] = v; __syncthreads();
        for (int st = 128; st > 0; st >>= 1) { if (t < st) red[t] += red[t + st]; __syncthreads(); }
        float mu = red[0] * (1.f / 256.f); __syncthreads();
        float d = v - mu;
        red[t] = d * d; __syncthreads();
        for (int st = 128; st > 0; st >>= 1) { if (t < st) red[t] += red[t + st]; __syncthreads(); }
        float var = red[0] * (1.f / 256.f); __syncthreads();
        prior_h[((size_t)b * 3 + pr) * C_DIM + t] =
            (_Float16)(d * rsqrtf(var + 1e-5f) * g[t] + bb[t]);
    }
}

// ---------------------------------------------------------------------------
// LN1 + cyclic shift + window partition -> xw (NW_TOT, 64, 256) f16
// ---------------------------------------------------------------------------
__global__ void k_ln1_window(const float* __restrict__ x,
                             const float* __restrict__ g, const float* __restrict__ bb,
                             _Float16* __restrict__ xw)
{
    __shared__ float red[256];
    const int blk = blockIdx.x;          // b*LTOK + h*112 + wcol
    const int c = threadIdx.x;
    float v = x[(size_t)blk * C_DIM + c];
    red[c] = v; __syncthreads();
    for (int s = 128; s > 0; s >>= 1) { if (c < s) red[c] += red[c + s]; __syncthreads(); }
    float mu = red[0] * (1.f / 256.f); __syncthreads();
    float d = v - mu;
    red[c] = d * d; __syncthreads();
    for (int s = 128; s > 0; s >>= 1) { if (c < s) red[c] += red[c + s]; __syncthreads(); }
    float var = red[0] * (1.f / 256.f);
    float nv = d * rsqrtf(var + 1e-5f) * g[c] + bb[c];

    int b = blk / LTOK, p = blk % LTOK;
    int h = p / HWDIM, wc = p % HWDIM;
    int sh = (h + HWDIM - SHIFT3) % HWDIM;
    int sw = (wc + HWDIM - SHIFT3) % HWDIM;
    int wh = sh / 7, ii = sh % 7, ww = sw / 7, jj = sw % 7;
    int win = b * NW_B + wh * 16 + ww;
    int n = ii * 7 + jj;
    xw[((size_t)win * 64 + n) * C_DIM + c] = (_Float16)nv;
}

// prior rows 49..51 per window, zero rows 52..63
__global__ void k_fill_prior(const _Float16* __restrict__ prior_h, _Float16* __restrict__ xw)
{
    const int w = blockIdx.x, c = threadIdx.x;
    const int b = w >> 8;
    for (int r = 49; r < 52; ++r)
        xw[((size_t)w * 64 + r) * C_DIM + c] = prior_h[((size_t)b * 3 + (r - 49)) * C_DIM + c];
    for (int r = 52; r < 64; ++r)
        xw[((size_t)w * 64 + r) * C_DIM + c] = (_Float16)0.f;
}

// ---------------------------------------------------------------------------
// QKV GEMM per window: (64x256) @ (768x256)^T -> (64x768), +bias, q*scale
// grid (6, NW_TOT)
// ---------------------------------------------------------------------------
__global__ void k_gemm_qkv(const _Float16* __restrict__ xw,
                           const _Float16* __restrict__ wh,
                           const float* __restrict__ bias,
                           _Float16* __restrict__ qkv)
{
    const int w = blockIdx.y, nb = blockIdx.x;
    v8f acc[4];
    wmma_block_64x128(xw + (size_t)w * 64 * C_DIM, C_DIM,
                      wh + (size_t)nb * 128 * C_DIM, C_DIM, C_DIM, acc);
    const int tid = threadIdx.x, lane = tid & 31, w8 = tid >> 5;
    const int mt = w8 >> 1, ntb = (w8 & 1) * 4;
#pragma unroll
    for (int t = 0; t < 4; ++t) {
        int ncol = nb * 128 + (ntb + t) * 16 + (lane & 15);
        float bsc = bias[ncol];
        float sc = (ncol < 256) ? QSCALE : 1.0f;
#pragma unroll
        for (int j = 0; j < 8; ++j) {
            int m = mt * 16 + (lane >> 4) * 8 + j;
            qkv[((size_t)w * 64 + m) * 768 + ncol] = (_Float16)((acc[t][j] + bsc) * sc);
        }
    }
}

// ---------------------------------------------------------------------------
// Attention per (head, window): scores WMMA -> bias+mask -> softmax -> P@V WMMA
// grid (NHEADS, NW_TOT)
// ---------------------------------------------------------------------------
__global__ void k_attn(const _Float16* __restrict__ qkv,
                       const float* __restrict__ rel_table,
                       const float* __restrict__ maskm,
                       _Float16* __restrict__ attn_out)
{
    __shared__ __attribute__((aligned(16))) _Float16 sQ[64 * 40];
    __shared__ __attribute__((aligned(16))) _Float16 sK[64 * 40];
    __shared__ float                                 sS[64 * 68];
    __shared__ __attribute__((aligned(16))) _Float16 sP[64 * 72];
    __shared__ __attribute__((aligned(16))) _Float16 sVT[32 * 72];

    const int head = blockIdx.x, w = blockIdx.y;
    const int tid = threadIdx.x, lane = tid & 31, w8 = tid >> 5;
    const _Float16* base = qkv + (size_t)w * 64 * 768;

    {   // stage Q (A) and K (B, token-major / K-contig) tiles
        int row = tid >> 2, off = (tid & 3) * 8;
        *(v8h*)&sQ[row * 40 + off] = *(const v8h*)&base[(size_t)row * 768 + head * DHEAD + off];
        *(v8h*)&sK[row * 40 + off] = *(const v8h*)&base[(size_t)row * 768 + 256 + head * DHEAD + off];
    }
    // stage V transposed [d][token], zero tokens >= 52
    for (int idx = tid; idx < 64 * 32; idx += 256) {
        int tok = idx >> 5, dd = idx & 31;
        _Float16 v = (tok < NKV) ? base[(size_t)tok * 768 + 512 + head * DHEAD + dd]
                                 : (_Float16)0.f;
        sVT[dd * 72 + tok] = v;
    }
    __syncthreads();

    // scores: 4x4 16x16 tiles, K=32 (one wmma each); wave -> mt=w8>>1, 2 n-tiles
    {
        const int mt = w8 >> 1, ntb = (w8 & 1) * 2;
        const v8f z = {0.f,0.f,0.f,0.f,0.f,0.f,0.f,0.f};
        v16h a = frag_a(&sQ[mt * 16 * 40], lane, 40);
#pragma unroll
        for (int t = 0; t < 2; ++t) {
            v16h b = frag_b(&sK[(ntb + t) * 16 * 40], lane, 40);
            v8f c = wmma_f16(a, b, z);
            int n = (ntb + t) * 16 + (lane & 15);
#pragma unroll
            for (int j = 0; j < 8; ++j) {
                int m = mt * 16 + (lane >> 4) * 8 + j;
                if (m < NTOK && n < NKV) {
                    float biasv;
                    if (n < NTOK) {
                        int im = m / 7, jm = m % 7, in_ = n / 7, jn = n % 7;
                        int idx = (im - in_ + 6) * 13 + (jm - jn + 6);
                        biasv = rel_table[idx * NHEADS + head]
                              + maskm[(((size_t)(w & 255)) * NTOK + m) * NTOK + n];
                    } else {
                        biasv = rel_table[(169 + n - NTOK) * NHEADS + head];
                    }
                    sS[m * 68 + n] = c[j] + biasv;
                }
            }
        }
    }
    __syncthreads();

    // softmax rows 0..48 over 52 cols; write P f16 (zero padding)
    if (tid < 64) {
        int m = tid;
        if (m < NTOK) {
            float mx = -1e30f;
            for (int n = 0; n < NKV; ++n) mx = fmaxf(mx, sS[m * 68 + n]);
            float sum = 0.f;
            for (int n = 0; n < NKV; ++n) sum += __expf(sS[m * 68 + n] - mx);
            float inv = 1.f / sum;
            for (int n = 0; n < 64; ++n)
                sP[m * 72 + n] = (n < NKV) ? (_Float16)(__expf(sS[m * 68 + n] - mx) * inv)
                                           : (_Float16)0.f;
        } else {
            for (int n = 0; n < 64; ++n) sP[m * 72 + n] = (_Float16)0.f;
        }
    }
    __syncthreads();

    // out = P(49x64pad) @ V(64x32): 4x2 tiles, 2 k-steps; wave -> one tile
    {
        const int mt = w8 >> 1, nt = w8 & 1;
        v8f c = {0.f,0.f,0.f,0.f,0.f,0.f,0.f,0.f};
#pragma unroll
        for (int kkt = 0; kkt < 2; ++kkt) {
            v16h a = frag_a(&sP[mt * 16 * 72 + kkt * 32], lane, 72);
            v16h b = frag_b(&sVT[nt * 16 * 72 + kkt * 32], lane, 72);
            c = wmma_f16(a, b, c);
        }
        int dd = nt * 16 + (lane & 15);
#pragma unroll
        for (int j = 0; j < 8; ++j) {
            int m = mt * 16 + (lane >> 4) * 8 + j;
            if (m < NTOK)
                attn_out[((size_t)w * NTOK + m) * C_DIM + head * DHEAD + dd] = (_Float16)c[j];
        }
    }
}

// ---------------------------------------------------------------------------
// Proj GEMM + bias + window-reverse + unshift + residual -> x_res f32
// grid (2, 784)
// ---------------------------------------------------------------------------
__global__ void k_gemm_proj(const _Float16* __restrict__ ao,
                            const _Float16* __restrict__ wh,
                            const float* __restrict__ bias,
                            const float* __restrict__ x,
                            float* __restrict__ xres)
{
    const int by = blockIdx.y, nb = blockIdx.x;
    v8f acc[4];
    wmma_block_64x128(ao + (size_t)by * 64 * C_DIM, C_DIM,
                      wh + (size_t)nb * 128 * C_DIM, C_DIM, C_DIM, acc);
    const int tid = threadIdx.x, lane = tid & 31, w8 = tid >> 5;
    const int mt = w8 >> 1, ntb = (w8 & 1) * 4;
#pragma unroll
    for (int t = 0; t < 4; ++t) {
        int ncol = nb * 128 + (ntb + t) * 16 + (lane & 15);
        float bsc = bias[ncol];
#pragma unroll
        for (int j = 0; j < 8; ++j) {
            int m = mt * 16 + (lane >> 4) * 8 + j;
            int r = by * 64 + m;                 // < 50176
            int w = r / NTOK, n49 = r % NTOK;
            int b = w >> 8, wi = w & 255;
            int whh = wi >> 4, www = wi & 15;
            int ii = n49 / 7, jj = n49 % 7;
            int h  = (whh * 7 + ii + SHIFT3) % HWDIM;
            int wc = (www * 7 + jj + SHIFT3) % HWDIM;
            size_t gidx = ((size_t)b * LTOK + h * HWDIM + wc) * C_DIM + ncol;
            xres[gidx] = x[gidx] + acc[t][j] + bsc;
        }
    }
}

// ---------------------------------------------------------------------------
// LN2: one wave per token
// ---------------------------------------------------------------------------
__global__ void k_ln2(const float* __restrict__ xr,
                      const float* __restrict__ g, const float* __restrict__ bb,
                      _Float16* __restrict__ outh)
{
    const int token = blockIdx.x * 8 + (threadIdx.x >> 5);
    const int lane = threadIdx.x & 31;
    const float* row = xr + (size_t)token * C_DIM;
    float v[8];
    float s = 0.f;
#pragma unroll
    for (int i = 0; i < 8; ++i) { v[i] = row[lane * 8 + i]; s += v[i]; }
    for (int m = 16; m > 0; m >>= 1) s += __shfl_xor(s, m, 32);
    float mu = s * (1.f / 256.f);
    float q = 0.f;
#pragma unroll
    for (int i = 0; i < 8; ++i) { float d = v[i] - mu; q += d * d; }
    for (int m = 16; m > 0; m >>= 1) q += __shfl_xor(q, m, 32);
    float rs = rsqrtf(q * (1.f / 256.f) + 1e-5f);
    _Float16* orow = outh + (size_t)token * C_DIM;
#pragma unroll
    for (int i = 0; i < 8; ++i) {
        int cc = lane * 8 + i;
        orow[cc] = (_Float16)((v[i] - mu) * rs * g[cc] + bb[cc]);
    }
}

// ---------------------------------------------------------------------------
// fc1 GEMM + bias + exact GELU -> f16 ; grid (8, 784)
// ---------------------------------------------------------------------------
__global__ void k_gemm_fc1(const _Float16* __restrict__ a,
                           const _Float16* __restrict__ wh,
                           const float* __restrict__ bias,
                           _Float16* __restrict__ hmid)
{
    const int by = blockIdx.y, nb = blockIdx.x;
    v8f acc[4];
    wmma_block_64x128(a + (size_t)by * 64 * C_DIM, C_DIM,
                      wh + (size_t)nb * 128 * C_DIM, C_DIM, C_DIM, acc);
    const int tid = threadIdx.x, lane = tid & 31, w8 = tid >> 5;
    const int mt = w8 >> 1, ntb = (w8 & 1) * 4;
#pragma unroll
    for (int t = 0; t < 4; ++t) {
        int ncol = nb * 128 + (ntb + t) * 16 + (lane & 15);
        float bsc = bias[ncol];
#pragma unroll
        for (int j = 0; j < 8; ++j) {
            int m = mt * 16 + (lane >> 4) * 8 + j;
            int r = by * 64 + m;
            float vv = acc[t][j] + bsc;
            float ge = 0.5f * vv * (1.f + erff(vv * 0.70710678118654752f));
            hmid[(size_t)r * HID + ncol] = (_Float16)ge;
        }
    }
}

// ---------------------------------------------------------------------------
// fc2 GEMM + bias + residual -> d_out f32 ; grid (2, 784)
// ---------------------------------------------------------------------------
__global__ void k_gemm_fc2(const _Float16* __restrict__ a,
                           const _Float16* __restrict__ wh,
                           const float* __restrict__ bias,
                           const float* __restrict__ xres,
                           float* __restrict__ out)
{
    const int by = blockIdx.y, nb = blockIdx.x;
    v8f acc[4];
    wmma_block_64x128(a + (size_t)by * 64 * HID, HID,
                      wh + (size_t)nb * 128 * HID, HID, HID, acc);
    const int tid = threadIdx.x, lane = tid & 31, w8 = tid >> 5;
    const int mt = w8 >> 1, ntb = (w8 & 1) * 4;
#pragma unroll
    for (int t = 0; t < 4; ++t) {
        int ncol = nb * 128 + (ntb + t) * 16 + (lane & 15);
        float bsc = bias[ncol];
#pragma unroll
        for (int j = 0; j < 8; ++j) {
            int m = mt * 16 + (lane >> 4) * 8 + j;
            int r = by * 64 + m;
            size_t gidx = (size_t)r * C_DIM + ncol;
            out[gidx] = xres[gidx] + acc[t][j] + bsc;
        }
    }
}

// ---------------------------------------------------------------------------
// Prior highway: v = (ph @ qkv_w.T + b)[v-part] ; out = v @ proj_w.T + proj_b
// grid 12 blocks (B*3 prior tokens) x 256
// ---------------------------------------------------------------------------
__global__ void k_vhw(const _Float16* __restrict__ prior_h,
                      const float* __restrict__ qkv_w, const float* __restrict__ qkv_b,
                      const float* __restrict__ proj_w, const float* __restrict__ proj_b,
                      float* __restrict__ out)
{
    __shared__ float vtok[256];
    const int tokidx = blockIdx.x;
    const int c = threadIdx.x;
    const _Float16* ph = prior_h + (size_t)tokidx * C_DIM;
    float s = qkv_b[512 + c];
    for (int k = 0; k < C_DIM; ++k) s += (float)ph[k] * qkv_w[(size_t)(512 + c) * C_DIM + k];
    vtok[c] = s;
    __syncthreads();
    float o = proj_b[c];
    for (int k = 0; k < C_DIM; ++k) o += vtok[k] * proj_w[(size_t)c * C_DIM + k];
    out[(size_t)tokidx * C_DIM + c] = o;
}

// ---------------------------------------------------------------------------
extern "C" void kernel_launch(void* const* d_in, const int* in_sizes, int n_in,
                              void* d_out, int out_size, void* d_ws, size_t ws_size,
                              hipStream_t stream)
{
    const float* x       = (const float*)d_in[0];
    const float* maskm   = (const float*)d_in[1];
    const float* area_fg = (const float*)d_in[2];
    const float* area_bg = (const float*)d_in[3];
    const float* area_uk = (const float*)d_in[4];
    const float* n1g     = (const float*)d_in[5];
    const float* n1b     = (const float*)d_in[6];
    const float* n2g     = (const float*)d_in[7];
    const float* n2b     = (const float*)d_in[8];
    const float* qkv_w   = (const float*)d_in[9];
    const float* qkv_b   = (const float*)d_in[10];
    const float* proj_w  = (const float*)d_in[11];
    const float* proj_b  = (const float*)d_in[12];
    const float* rel_t   = (const float*)d_in[13];
    const float* fc1_w   = (const float*)d_in[14];
    const float* fc1_b   = (const float*)d_in[15];
    const float* fc2_w   = (const float*)d_in[16];
    const float* fc2_b   = (const float*)d_in[17];

    float* out_x  = (float*)d_out;
    float* out_vh = out_x + (size_t)BATCH * LTOK * C_DIM;

    // workspace carve-up (256B aligned)
    char* p = (char*)d_ws;
    auto carve = [&](size_t bytes) -> char* {
        char* r = p; p += (bytes + 255) & ~(size_t)255; return r;
    };
    _Float16* qkvw_h  = (_Float16*)carve((size_t)768  * 256  * 2);
    _Float16* projw_h = (_Float16*)carve((size_t)256  * 256  * 2);
    _Float16* fc1w_h  = (_Float16*)carve((size_t)1024 * 256  * 2);
    _Float16* fc2w_h  = (_Float16*)carve((size_t)256  * 1024 * 2);
    _Float16* prior_h = (_Float16*)carve((size_t)12 * 256 * 2);
    _Float16* xw_h    = (_Float16*)carve((size_t)NW_TOT * 64 * C_DIM * 2);
    _Float16* qkv_h   = (_Float16*)carve((size_t)NW_TOT * 64 * 768 * 2);
    _Float16* ao_h    = (_Float16*)carve((size_t)MROWS * C_DIM * 2);
    float*    xres    = (float*)   carve((size_t)MROWS * C_DIM * 4);
    _Float16* ln2_h   = (_Float16*)carve((size_t)MROWS * C_DIM * 2);
    _Float16* hmid_h  = (_Float16*)carve((size_t)MROWS * HID * 2);

    // weight conversion
    k_cvt<<<(768 * 256 + 255) / 256, 256, 0, stream>>>(qkv_w, qkvw_h, 768 * 256);
    k_cvt<<<(256 * 256 + 255) / 256, 256, 0, stream>>>(proj_w, projw_h, 256 * 256);
    k_cvt<<<(1024 * 256 + 255) / 256, 256, 0, stream>>>(fc1_w, fc1w_h, 1024 * 256);
    k_cvt<<<(256 * 1024 + 255) / 256, 256, 0, stream>>>(fc2_w, fc2w_h, 256 * 1024);

    // priors + LN
    k_priors<<<BATCH, 256, 0, stream>>>(x, area_uk, area_fg, area_bg, n1g, n1b, prior_h);

    // LN1 + shift + window partition
    k_ln1_window<<<BATCH * LTOK, 256, 0, stream>>>(x, n1g, n1b, xw_h);
    k_fill_prior<<<NW_TOT, 256, 0, stream>>>(prior_h, xw_h);

    // QKV
    k_gemm_qkv<<<dim3(6, NW_TOT), 256, 0, stream>>>(xw_h, qkvw_h, qkv_b, qkv_h);

    // attention
    k_attn<<<dim3(NHEADS, NW_TOT), 256, 0, stream>>>(qkv_h, rel_t, maskm, ao_h);

    // proj + reverse window/shift + residual
    k_gemm_proj<<<dim3(2, MROWS / 64), 256, 0, stream>>>(ao_h, projw_h, proj_b, x, xres);

    // LN2
    k_ln2<<<MROWS / 8, 256, 0, stream>>>(xres, n2g, n2b, ln2_h);

    // MLP
    k_gemm_fc1<<<dim3(8, MROWS / 64), 256, 0, stream>>>(ln2_h, fc1w_h, fc1_b, hmid_h);
    k_gemm_fc2<<<dim3(2, MROWS / 64), 256, 0, stream>>>(hmid_h, fc2w_h, fc2_b, xres, out_x);

    // prior highway output
    k_vhw<<<12, 256, 0, stream>>>(prior_h, qkv_w, qkv_b, proj_w, proj_b, out_vh);
}
